// GAT2_84361747628050
// MI455X (gfx1250) — matrix operand order repaired and checked by reading, not compiled, not run on hardware
//
#include <hip/hip_runtime.h>
#include <hip/hip_bf16.h>
#include <stdint.h>

// ---------------------------------------------------------------------------
// Problem constants (from the reference)
// ---------------------------------------------------------------------------
#define NNODES 50000
#define NEDGES 312500
#define ETOT   (NEDGES + NNODES)   // edges + self loops
#define DIN    2613
#define KPAD1  2624                // DIN padded to multiple of 32
#define DHID   256
#define BN_EPS 1e-5f
#define NEG_SLOPE 0.2f

typedef __attribute__((ext_vector_type(16))) __bf16 v16bf;
typedef __attribute__((ext_vector_type(8)))  float  v8f;

union ABfrag {
    v16bf    v;
    __bf16   e[16];
    uint32_t u[8];
    uint4    q[2];
};

// CDNA5 async global->LDS staging (ASYNCcnt path). The builtin takes
// v4i pointers in addrspace(1)/(3) (b128 payload), per the clang signature.
#if defined(__HIP_DEVICE_COMPILE__) && \
    __has_builtin(__builtin_amdgcn_global_load_async_to_lds_b128)
#define USE_ASYNC_LDS 1
typedef int async_v4i __attribute__((vector_size(16)));
#define AS_GLOBAL(p) ((__attribute__((address_space(1))) async_v4i*)(p))
#define AS_LOCAL(p)  ((__attribute__((address_space(3))) async_v4i*)(p))
#else
#define USE_ASYNC_LDS 0
#endif

// ---------------------------------------------------------------------------
// Weight prep: W[K x 256] fp32 -> transposed [256 x KPAD] bf16 hi + residual lo
// (K-pairs dword-contiguous for the WMMA B fragment loads; zero padded tail)
// ---------------------------------------------------------------------------
__global__ void k_conv_weight(const float* __restrict__ W,
                              __bf16* __restrict__ Whi, __bf16* __restrict__ Wlo,
                              int K, int KPAD)
{
    int idx = blockIdx.x * blockDim.x + threadIdx.x;
    if (idx >= DHID * KPAD) return;
    int n = idx / KPAD;
    int k = idx - n * KPAD;
    float w = (k < K) ? W[(size_t)k * DHID + n] : 0.0f;
    __bf16 h = (__bf16)w;
    Whi[(size_t)n * KPAD + k] = h;
    Wlo[(size_t)n * KPAD + k] = (__bf16)(w - (float)h);
}

// ---------------------------------------------------------------------------
// Split-bf16 GEMM:  C[M x 256] = A[M x K] (fp32) * W[K x 256]
// W supplied transposed as bf16 hi/lo ([256][KPAD]).
// Per 32-K step each wave issues 3 v_wmma_f32_16x16x32_bf16 per N-tile:
//   Ahi*Bhi + Ahi*Blo + Alo*Bhi  (~fp32 accuracy at bf16 matrix-pipe rate).
// Double-buffered LDS staging of the 32K x 256N W slice, filled through the
// gfx1250 async global->LDS path (overlaps with the WMMA stream, ASYNCcnt).
// Block = 256 threads = 8 wave32s = 4 M-tiles x 2 column halves.
// ---------------------------------------------------------------------------
__global__ __launch_bounds__(256)
void k_gemm_bf16x3(const float* __restrict__ A,
                   const __bf16* __restrict__ Bhi, const __bf16* __restrict__ Blo,
                   float* __restrict__ C, int M, int K, int KPAD)
{
    // Row pitch 20 dwords keeps b128 16-B alignment while breaking the 64-bank
    // stride pattern. 2 buffers x (hi+lo) x 20KB = 80KB of the 320KB WGP LDS.
    __shared__ uint32_t ldsHi[2][DHID * 20];
    __shared__ uint32_t ldsLo[2][DHID * 20];

    const int tid  = threadIdx.x;
    const int lane = tid & 31;
    const int wave = tid >> 5;
    const int mt   = blockIdx.x * 4 + (wave >> 1);
    const int ch   = wave & 1;              // column half: 0 -> n 0..127, 1 -> 128..255
    const int m0   = mt * 16;
    const bool active = (m0 < M);

    const int aKoff = (lane < 16) ? 0 : 16; // A fragment K sub-offset (lane<16: K 0..15)
    const int bBase = (lane < 16) ? 0 : 8;  // B fragment dword offset (lane<16: K 0..15)
    const float* Arow = A + (size_t)(m0 + (lane & 15)) * (size_t)K;

    // Stage one 32-K slice (thread t copies 16 dwords = 32 bf16 of column t).
    auto stage = [&](int buf, int kk) {
#if USE_ASYNC_LDS
        const char* gh = (const char*)(Bhi + (size_t)tid * KPAD + kk);
        const char* gl = (const char*)(Blo + (size_t)tid * KPAD + kk);
        uint32_t*   dh = &ldsHi[buf][tid * 20];
        uint32_t*   dl = &ldsLo[buf][tid * 20];
        __builtin_amdgcn_global_load_async_to_lds_b128(AS_GLOBAL(gh +  0), AS_LOCAL(dh +  0), 0, 0);
        __builtin_amdgcn_global_load_async_to_lds_b128(AS_GLOBAL(gh + 16), AS_LOCAL(dh +  4), 0, 0);
        __builtin_amdgcn_global_load_async_to_lds_b128(AS_GLOBAL(gh + 32), AS_LOCAL(dh +  8), 0, 0);
        __builtin_amdgcn_global_load_async_to_lds_b128(AS_GLOBAL(gh + 48), AS_LOCAL(dh + 12), 0, 0);
        __builtin_amdgcn_global_load_async_to_lds_b128(AS_GLOBAL(gl +  0), AS_LOCAL(dl +  0), 0, 0);
        __builtin_amdgcn_global_load_async_to_lds_b128(AS_GLOBAL(gl + 16), AS_LOCAL(dl +  4), 0, 0);
        __builtin_amdgcn_global_load_async_to_lds_b128(AS_GLOBAL(gl + 32), AS_LOCAL(dl +  8), 0, 0);
        __builtin_amdgcn_global_load_async_to_lds_b128(AS_GLOBAL(gl + 48), AS_LOCAL(dl + 12), 0, 0);
#else
        const uint4* sh = (const uint4*)(Bhi + (size_t)tid * KPAD + kk);
        const uint4* sl = (const uint4*)(Blo + (size_t)tid * KPAD + kk);
        uint4* dh = (uint4*)&ldsHi[buf][tid * 20];
        uint4* dl = (uint4*)&ldsLo[buf][tid * 20];
        #pragma unroll
        for (int q = 0; q < 4; ++q) { dh[q] = sh[q]; dl[q] = sl[q]; }
#endif
    };
    auto wait_stage = [&]() {
#if USE_ASYNC_LDS
        asm volatile("s_wait_asynccnt 0x0" ::: "memory");
#endif
    };

    v8f acc[8];
    const v8f vzero = {0.f, 0.f, 0.f, 0.f, 0.f, 0.f, 0.f, 0.f};
    #pragma unroll
    for (int t = 0; t < 8; ++t) acc[t] = vzero;

    // Pipeline prologue
    stage(0, 0);
    wait_stage();
    __syncthreads();

    int cur = 0;
    for (int k0 = 0; k0 < KPAD; k0 += 32) {
        if (k0 + 32 < KPAD) stage(cur ^ 1, k0 + 32);   // overlaps with WMMAs below

        if (active) {
            if (k0 + 64 <= K) __builtin_prefetch(&Arow[k0 + 32], 0, 1);

            // A fragment: fp32 -> bf16 hi + residual lo, WMMA A layout.
            ABfrag ahi, alo;
            #pragma unroll
            for (int i = 0; i < 16; ++i) {
                int j = i >> 1;
                int k = k0 + aKoff + i; (void)j;
                float av = (k < K) ? Arow[k] : 0.0f;
                __bf16 h = (__bf16)av;
                ahi.e[i] = h;
                alo.e[i] = (__bf16)(av - (float)h);
            }

            const uint32_t* bufH = ldsHi[cur];
            const uint32_t* bufL = ldsLo[cur];
            // Pairwise-interleaved N-tiles: consecutive WMMAs never share an
            // accumulator, breaking D->C chains / WAR stalls on B registers.
            #pragma unroll
            for (int nt = 0; nt < 8; nt += 2) {
                const int n0 = ch * 128 + nt * 16 + (lane & 15);
                const int n1 = n0 + 16;
                ABfrag bh0, bl0, bh1, bl1;
                bh0.q[0] = *(const uint4*)&bufH[n0 * 20 + bBase];
                bh0.q[1] = *(const uint4*)&bufH[n0 * 20 + bBase + 4];
                bl0.q[0] = *(const uint4*)&bufL[n0 * 20 + bBase];
                bl0.q[1] = *(const uint4*)&bufL[n0 * 20 + bBase + 4];
                bh1.q[0] = *(const uint4*)&bufH[n1 * 20 + bBase];
                bh1.q[1] = *(const uint4*)&bufH[n1 * 20 + bBase + 4];
                bl1.q[0] = *(const uint4*)&bufL[n1 * 20 + bBase];
                bl1.q[1] = *(const uint4*)&bufL[n1 * 20 + bBase + 4];
                acc[nt]     = __builtin_amdgcn_wmma_f32_16x16x32_bf16(
                    false, ahi.v, false, bh0.v, (short)0, acc[nt],     false, false);
                acc[nt + 1] = __builtin_amdgcn_wmma_f32_16x16x32_bf16(
                    false, ahi.v, false, bh1.v, (short)0, acc[nt + 1], false, false);
                acc[nt]     = __builtin_amdgcn_wmma_f32_16x16x32_bf16(
                    false, ahi.v, false, bl0.v, (short)0, acc[nt],     false, false);
                acc[nt + 1] = __builtin_amdgcn_wmma_f32_16x16x32_bf16(
                    false, ahi.v, false, bl1.v, (short)0, acc[nt + 1], false, false);
                acc[nt]     = __builtin_amdgcn_wmma_f32_16x16x32_bf16(
                    false, alo.v, false, bh0.v, (short)0, acc[nt],     false, false);
                acc[nt + 1] = __builtin_amdgcn_wmma_f32_16x16x32_bf16(
                    false, alo.v, false, bh1.v, (short)0, acc[nt + 1], false, false);
            }
        }

        wait_stage();      // our async stage of the next slice has landed
        __syncthreads();   // everyone done reading cur / staging next
        cur ^= 1;
    }

    if (active) {
        const int rBase = (lane < 16) ? 0 : 8;   // C layout: VGPR r -> M=r / M=8+r
        #pragma unroll
        for (int nt = 0; nt < 8; ++nt) {
            const int col = ch * 128 + nt * 16 + (lane & 15);
            #pragma unroll
            for (int r = 0; r < 8; ++r) {
                const int row = m0 + rBase + r;
                if (row < M) C[(size_t)row * DHID + col] = acc[nt][r];
            }
        }
    }
}

// ---------------------------------------------------------------------------
// BatchNorm (training stats) + ELU.  Linear biases are omitted: BN subtracts
// the batch mean, so a per-feature bias cancels exactly (and they are zero).
// ---------------------------------------------------------------------------
#define BN_ROWS 256
#define BN_BLKS ((NNODES + BN_ROWS - 1) / BN_ROWS)   // 196

__global__ void k_bn_partial(const float* __restrict__ h,
                             float* __restrict__ psum, float* __restrict__ psq)
{
    const int t  = threadIdx.x;           // feature
    const int r0 = blockIdx.x * BN_ROWS;
    const int r1 = (r0 + BN_ROWS < NNODES) ? (r0 + BN_ROWS) : NNODES;
    float s = 0.f, q = 0.f;
    for (int r = r0; r < r1; ++r) {
        float v = h[(size_t)r * DHID + t]; // consecutive lanes -> consecutive features
        s += v; q += v * v;
    }
    psum[blockIdx.x * DHID + t] = s;
    psq [blockIdx.x * DHID + t] = q;
}

__global__ void k_bn_finalize(const float* __restrict__ psum, const float* __restrict__ psq,
                              const float* __restrict__ gamma, const float* __restrict__ beta,
                              float* __restrict__ scale, float* __restrict__ shift)
{
    const int t = threadIdx.x;
    float s = 0.f, q = 0.f;
    for (int b = 0; b < BN_BLKS; ++b) { s += psum[b * DHID + t]; q += psq[b * DHID + t]; }
    const float inv_n = 1.0f / (float)NNODES;
    float mean = s * inv_n;
    float var  = q * inv_n - mean * mean;
    float sc   = gamma[t] * rsqrtf(var + BN_EPS);
    scale[t] = sc;
    shift[t] = beta[t] - mean * sc;
}

__global__ void k_bn_apply_elu(const float* __restrict__ in,
                               const float* __restrict__ scale, const float* __restrict__ shift,
                               float* __restrict__ out)
{
    size_t i = (size_t)blockIdx.x * blockDim.x + threadIdx.x;
    if (i >= (size_t)NNODES * DHID) return;
    int f = (int)(i & (DHID - 1));
    float v = in[i] * scale[f] + shift[f];
    out[i] = (v > 0.f) ? v : (__expf(v) - 1.f);
}

// ---------------------------------------------------------------------------
// GAT pieces
// ---------------------------------------------------------------------------
__global__ void k_build_edges(const int* __restrict__ esrc, const int* __restrict__ edst,
                              int* __restrict__ srcs, int* __restrict__ dsts)
{
    int i = blockIdx.x * blockDim.x + threadIdx.x;
    if (i < NEDGES)     { srcs[i] = esrc[i]; dsts[i] = edst[i]; }
    else if (i < ETOT)  { srcs[i] = i - NEDGES; dsts[i] = i - NEDGES; }  // self loops
}

// Per-node attention dots (wave32 per node, shuffle reduction)
__global__ void k_att_dots(const float* __restrict__ h,
                           const float* __restrict__ av_s, const float* __restrict__ av_d,
                           float* __restrict__ a_s, float* __restrict__ a_d)
{
    const int gw   = (blockIdx.x * blockDim.x + threadIdx.x) >> 5;
    const int lane = threadIdx.x & 31;
    if (gw >= NNODES) return;
    const float* row = h + (size_t)gw * DHID;
    float s = 0.f, d = 0.f;
    #pragma unroll
    for (int i = 0; i < DHID; i += 32) {
        float v = row[i + lane];
        s += v * av_s[i + lane];
        d += v * av_d[i + lane];
    }
    #pragma unroll
    for (int m = 16; m >= 1; m >>= 1) {
        s += __shfl_xor(s, m, 32);
        d += __shfl_xor(d, m, 32);
    }
    if (lane == 0) { a_s[gw] = s; a_d[gw] = d; }
}

__device__ __forceinline__ unsigned f32_to_ordered(float f) {
    unsigned b = __float_as_uint(f);
    return (b & 0x80000000u) ? ~b : (b | 0x80000000u);
}
__device__ __forceinline__ float ordered_to_f32(unsigned k) {
    unsigned b = (k & 0x80000000u) ? (k ^ 0x80000000u) : ~k;
    return __uint_as_float(b);
}

__global__ void k_softmax_init(unsigned* __restrict__ emax, float* __restrict__ denom)
{
    int i = blockIdx.x * blockDim.x + threadIdx.x;
    if (i < NNODES) { emax[i] = f32_to_ordered(-__builtin_inff()); denom[i] = 0.f; }
}

__global__ void k_edge_logits(const int* __restrict__ srcs, const int* __restrict__ dsts,
                              const float* __restrict__ a_s, const float* __restrict__ a_d,
                              float* __restrict__ e_val, unsigned* __restrict__ emax)
{
    int i = blockIdx.x * blockDim.x + threadIdx.x;
    if (i >= ETOT) return;
    float x = a_s[srcs[i]] + a_d[dsts[i]];
    float e = (x > 0.f) ? x : NEG_SLOPE * x;     // leaky_relu
    e_val[i] = e;
    atomicMax(&emax[dsts[i]], f32_to_ordered(e));
}

__global__ void k_edge_exp(const int* __restrict__ dsts,
                           const float* __restrict__ e_val, const unsigned* __restrict__ emax,
                           float* __restrict__ e_exp, float* __restrict__ denom)
{
    int i = blockIdx.x * blockDim.x + threadIdx.x;
    if (i >= ETOT) return;
    int d = dsts[i];
    float ex = __expf(e_val[i] - ordered_to_f32(emax[d]));
    e_exp[i] = ex;
    atomicAdd(&denom[d], ex);
}

// Scatter: out[dst] += alpha_e * h[src]   (wave32 per edge, 8 feats/lane)
__global__ void k_gat_agg(const int* __restrict__ srcs, const int* __restrict__ dsts,
                          const float* __restrict__ e_exp, const float* __restrict__ denom,
                          const float* __restrict__ h, float* __restrict__ out)
{
    const int gw   = (blockIdx.x * blockDim.x + threadIdx.x) >> 5;
    const int lane = threadIdx.x & 31;
    if (gw >= ETOT) return;
    const int s = srcs[gw], d = dsts[gw];
    const float alpha = e_exp[gw] / (denom[d] + 1e-16f);
    const float* hr = h + (size_t)s * DHID;
    float* orow     = out + (size_t)d * DHID;
    #pragma unroll
    for (int i = 0; i < DHID; i += 32)
        atomicAdd(&orow[i + lane], alpha * hr[i + lane]);
}

__global__ void k_bias_elu(const float* __restrict__ in, const float* __restrict__ bias,
                           float* __restrict__ out)
{
    size_t i = (size_t)blockIdx.x * blockDim.x + threadIdx.x;
    if (i >= (size_t)NNODES * DHID) return;
    float v = in[i] + bias[i & (DHID - 1)];
    out[i] = (v > 0.f) ? v : (__expf(v) - 1.f);
}

__global__ void k_zero_f32(float* __restrict__ p, size_t n)
{
    size_t i = (size_t)blockIdx.x * blockDim.x + threadIdx.x;
    if (i < n) p[i] = 0.f;
}

// ---------------------------------------------------------------------------
// Host orchestration
// ---------------------------------------------------------------------------
namespace {

struct Ws {
    float *ws0, *ws1;
    __bf16 *W1hi, *W1lo, *W2hi, *W2lo, *W3hi, *W3lo, *Wg1hi, *Wg1lo, *Wg2hi, *Wg2lo;
    int *srcs, *dsts;
    float *a_s, *a_d, *e_val, *e_exp, *denom;
    unsigned *emax;
    float *psum, *psq, *scale, *shift;
};

inline void* carve(char*& p, size_t bytes) {
    void* r = p;
    p += (bytes + 255) & ~(size_t)255;
    return r;
}

inline void run_gemm(const float* A, const __bf16* Whi, const __bf16* Wlo, float* C,
                     int M, int K, int KPAD, hipStream_t s)
{
    int mtiles = (M + 15) / 16;
    int blocks = (mtiles + 3) / 4;
    k_gemm_bf16x3<<<blocks, 256, 0, s>>>(A, Whi, Wlo, C, M, K, KPAD);
}

inline void run_bn_elu(const float* in, const float* g, const float* be, float* out,
                       const Ws& w, hipStream_t s)
{
    k_bn_partial <<<BN_BLKS, DHID, 0, s>>>(in, w.psum, w.psq);
    k_bn_finalize<<<1, DHID, 0, s>>>(w.psum, w.psq, g, be, w.scale, w.shift);
    size_t total = (size_t)NNODES * DHID;
    k_bn_apply_elu<<<(unsigned)((total + 255) / 256), 256, 0, s>>>(in, w.scale, w.shift, out);
}

inline void run_gat(const float* h_in, const __bf16* Whi, const __bf16* Wlo,
                    const float* att_s, const float* att_d, const float* bias,
                    float* hproj /*scratch*/, float* agg /*accum*/, float* out,
                    const Ws& w, hipStream_t s)
{
    const size_t total = (size_t)NNODES * DHID;
    run_gemm(h_in, Whi, Wlo, hproj, NNODES, DHID, DHID, s);
    k_att_dots<<<(NNODES * 32 + 255) / 256, 256, 0, s>>>(hproj, att_s, att_d, w.a_s, w.a_d);
    k_softmax_init<<<(NNODES + 255) / 256, 256, 0, s>>>(w.emax, w.denom);
    k_zero_f32<<<(unsigned)((total + 255) / 256), 256, 0, s>>>(agg, total);
    k_edge_logits<<<(ETOT + 255) / 256, 256, 0, s>>>(w.srcs, w.dsts, w.a_s, w.a_d, w.e_val, w.emax);
    k_edge_exp   <<<(ETOT + 255) / 256, 256, 0, s>>>(w.dsts, w.e_val, w.emax, w.e_exp, w.denom);
    k_gat_agg<<<((size_t)ETOT * 32 + 255) / 256, 256, 0, s>>>(w.srcs, w.dsts, w.e_exp, w.denom, hproj, agg);
    k_bias_elu<<<(unsigned)((total + 255) / 256), 256, 0, s>>>(agg, bias, out);
}

} // namespace

extern "C" void kernel_launch(void* const* d_in, const int* in_sizes, int n_in,
                              void* d_out, int out_size, void* d_ws, size_t ws_size,
                              hipStream_t stream)
{
    (void)in_sizes; (void)n_in; (void)out_size; (void)ws_size;

    // Input order: x, edges, W1,b1,g1,be1, W2,b2,g2,be2, W3,b3,g3,be3,
    //              Wg1,as1,ad1,bg1, Wg2,as2,ad2,bg2
    const float* x     = (const float*)d_in[0];
    const int*   edges = (const int*)  d_in[1];
    const float* W1  = (const float*)d_in[2];
    const float* g1  = (const float*)d_in[4];
    const float* be1 = (const float*)d_in[5];
    const float* W2  = (const float*)d_in[6];
    const float* g2  = (const float*)d_in[8];
    const float* be2 = (const float*)d_in[9];
    const float* W3  = (const float*)d_in[10];
    const float* g3  = (const float*)d_in[12];
    const float* be3 = (const float*)d_in[13];
    const float* Wg1 = (const float*)d_in[14];
    const float* as1 = (const float*)d_in[15];
    const float* ad1 = (const float*)d_in[16];
    const float* bg1 = (const float*)d_in[17];
    const float* Wg2 = (const float*)d_in[18];
    const float* as2 = (const float*)d_in[19];
    const float* ad2 = (const float*)d_in[20];
    const float* bg2 = (const float*)d_in[21];
    float* out = (float*)d_out;

    // ---- workspace carve (~112 MB) ----
    char* p = (char*)d_ws;
    Ws w;
    const size_t hbytes = (size_t)NNODES * DHID * sizeof(float);
    w.ws0   = (float*)carve(p, hbytes);
    w.ws1   = (float*)carve(p, hbytes);
    w.W1hi  = (__bf16*)carve(p, (size_t)DHID * KPAD1 * 2);
    w.W1lo  = (__bf16*)carve(p, (size_t)DHID * KPAD1 * 2);
    w.W2hi  = (__bf16*)carve(p, (size_t)DHID * DHID * 2);
    w.W2lo  = (__bf16*)carve(p, (size_t)DHID * DHID * 2);
    w.W3hi  = (__bf16*)carve(p, (size_t)DHID * DHID * 2);
    w.W3lo  = (__bf16*)carve(p, (size_t)DHID * DHID * 2);
    w.Wg1hi = (__bf16*)carve(p, (size_t)DHID * DHID * 2);
    w.Wg1lo = (__bf16*)carve(p, (size_t)DHID * DHID * 2);
    w.Wg2hi = (__bf16*)carve(p, (size_t)DHID * DHID * 2);
    w.Wg2lo = (__bf16*)carve(p, (size_t)DHID * DHID * 2);
    w.srcs  = (int*)carve(p, (size_t)ETOT * 4);
    w.dsts  = (int*)carve(p, (size_t)ETOT * 4);
    w.a_s   = (float*)carve(p, (size_t)NNODES * 4);
    w.a_d   = (float*)carve(p, (size_t)NNODES * 4);
    w.e_val = (float*)carve(p, (size_t)ETOT * 4);
    w.e_exp = (float*)carve(p, (size_t)ETOT * 4);
    w.denom = (float*)carve(p, (size_t)NNODES * 4);
    w.emax  = (unsigned*)carve(p, (size_t)NNODES * 4);
    w.psum  = (float*)carve(p, (size_t)BN_BLKS * DHID * 4);
    w.psq   = (float*)carve(p, (size_t)BN_BLKS * DHID * 4);
    w.scale = (float*)carve(p, DHID * 4);
    w.shift = (float*)carve(p, DHID * 4);

    // ---- weight prep (bf16 hi/lo, transposed, padded) ----
    {
        int tot1 = DHID * KPAD1, totS = DHID * DHID;
        k_conv_weight<<<(tot1 + 255) / 256, 256, 0, stream>>>(W1,  w.W1hi,  w.W1lo,  DIN,  KPAD1);
        k_conv_weight<<<(totS + 255) / 256, 256, 0, stream>>>(W2,  w.W2hi,  w.W2lo,  DHID, DHID);
        k_conv_weight<<<(totS + 255) / 256, 256, 0, stream>>>(W3,  w.W3hi,  w.W3lo,  DHID, DHID);
        k_conv_weight<<<(totS + 255) / 256, 256, 0, stream>>>(Wg1, w.Wg1hi, w.Wg1lo, DHID, DHID);
        k_conv_weight<<<(totS + 255) / 256, 256, 0, stream>>>(Wg2, w.Wg2hi, w.Wg2lo, DHID, DHID);
    }
    // ---- edge list with self loops ----
    k_build_edges<<<(ETOT + 255) / 256, 256, 0, stream>>>(edges, edges + NEDGES, w.srcs, w.dsts);

    // ---- MLP: 3x (linear -> batchnorm -> elu); biases cancel inside BN ----
    run_gemm(x, w.W1hi, w.W1lo, w.ws0, NNODES, DIN, KPAD1, stream);
    run_bn_elu(w.ws0, g1, be1, w.ws1, w, stream);            // h1 -> ws1
    run_gemm(w.ws1, w.W2hi, w.W2lo, w.ws0, NNODES, DHID, DHID, stream);
    run_bn_elu(w.ws0, g2, be2, w.ws1, w, stream);            // h2 -> ws1
    run_gemm(w.ws1, w.W3hi, w.W3lo, w.ws0, NNODES, DHID, DHID, stream);
    run_bn_elu(w.ws0, g3, be3, w.ws1, w, stream);            // h3 -> ws1

    // ---- GATConv 1: ws1 -> ws1 (agg staged in d_out, elu'd into ws1) ----
    run_gat(w.ws1, w.Wg1hi, w.Wg1lo, as1, ad1, bg1, w.ws0, out, w.ws1, w, stream);
    // ---- GATConv 2: ws1 -> d_out ----
    run_gat(w.ws1, w.Wg2hi, w.Wg2lo, as2, ad2, bg2, w.ws0, out, out, w, stream);
}
